// Decoder_PAC_67946382622909
// MI455X (gfx1250) — matrix operand order repaired and checked by reading, not compiled, not run on hardware
//
#include <hip/hip_runtime.h>
#include <hip/hip_bf16.h>
#include <math.h>

// ---------------------------------------------------------------------------
// CDNA5 (gfx1250) wave32 WMMA types
// ---------------------------------------------------------------------------
typedef __attribute__((ext_vector_type(16))) __bf16        v16bf;
typedef __attribute__((ext_vector_type(8)))  float         v8f;
typedef __attribute__((ext_vector_type(8)))  unsigned int  v8u;

// A-matrix (16x32 bf16) K index for vector element t (0..15), lane half (0/1).
// ISA 7.12.2: lanes 0-15 -> K {0..7,16..23}, lanes 16-31 -> K {8..15,24..31}.
__device__ __forceinline__ int kmapA(int t, int half) {
  int i = t >> 1, e = t & 1;
  int base = (i < 4) ? (2 * i) : (16 + 2 * (i - 4));
  return base + 8 * half + e;
}

__device__ __forceinline__ unsigned int pk_mul_bf16(unsigned int a, unsigned int b) {
  unsigned int d;
  asm("v_pk_mul_bf16 %0, %1, %2" : "=v"(d) : "v"(a), "v"(b));
  return d;
}

__device__ __forceinline__ unsigned int bf16pair(float f) {
  unsigned short u = __builtin_bit_cast(unsigned short, (__bf16)f);
  return ((unsigned int)u << 16) | u;
}

// ---------------------------------------------------------------------------
// Instance norm + residual (per channel), optional bf16 NHWC side output.
// ---------------------------------------------------------------------------
__global__ void __launch_bounds__(256)
inorm_res_kernel(const float* __restrict__ x, float* __restrict__ y,
                 __bf16* __restrict__ yb, int C, int HW) {
  const int c = blockIdx.x;
  const float* xc = x + (size_t)c * HW;
  float s = 0.f, s2 = 0.f;
  for (int i = threadIdx.x; i < HW; i += 256) {
    float v = xc[i];
    s += v; s2 += v * v;
  }
  __shared__ float red0[256], red1[256];
  red0[threadIdx.x] = s; red1[threadIdx.x] = s2;
  __syncthreads();
  for (int off = 128; off > 0; off >>= 1) {
    if (threadIdx.x < off) {
      red0[threadIdx.x] += red0[threadIdx.x + off];
      red1[threadIdx.x] += red1[threadIdx.x + off];
    }
    __syncthreads();
  }
  float mean = red0[0] / (float)HW;
  float var  = red1[0] / (float)HW - mean * mean;
  float rs   = rsqrtf(var + 1e-5f);
  float* yc = y + (size_t)c * HW;
  for (int i = threadIdx.x; i < HW; i += 256) {
    float v = xc[i];
    float r = (v - mean) * rs + v;
    yc[i] = r;
    if (yb) yb[(size_t)i * C + c] = (__bf16)r;
  }
}

// ---------------------------------------------------------------------------
// NCHW f32 -> NHWC bf16 staging
// ---------------------------------------------------------------------------
__global__ void __launch_bounds__(256)
nchw_to_nhwc_bf16(const float* __restrict__ in, __bf16* __restrict__ outb,
                  int C, int HW) {
  long long idx = (long long)blockIdx.x * blockDim.x + threadIdx.x;
  if (idx >= (long long)C * HW) return;
  int c = (int)(idx / HW), p = (int)(idx - (long long)c * HW);
  outb[(size_t)p * C + c] = (__bf16)in[idx];
}

// ---------------------------------------------------------------------------
// Weight pre-pack into WMMA A-fragment image:
//   wp[coTile][ciChunk][tap][lane][pair]  (uint = bf16 pair), co >= Cout -> 0
// transposed=0: w is OIHW (conv);  transposed=1: w is (Cin,Cout,3,3) (PAC)
// ---------------------------------------------------------------------------
__global__ void __launch_bounds__(256)
prepack_wmma_A(const float* __restrict__ w, unsigned int* __restrict__ wp,
               int Cin, int Cout, int coutTiles, int transposed) {
  int idx = blockIdx.x * blockDim.x + threadIdx.x;
  int KC = Cin >> 5;
  int total = coutTiles * KC * 9 * 256;
  if (idx >= total) return;
  int pair = idx & 7;
  int lane = (idx >> 3) & 31;
  int rest = idx >> 8;                 // ((coTile*KC)+ciChunk)*9 + tap
  int tap = rest % 9;
  int rest2 = rest / 9;
  int ciChunk = rest2 % KC;
  int coTile = rest2 / KC;
  int half = lane >> 4, m = lane & 15;
  int co = coTile * 16 + m;
  int ci = (ciChunk << 5) + 16 * half + pair * 2;
  float f0 = 0.f, f1 = 0.f;
  if (co < Cout) {
    if (transposed) {
      f0 = w[((size_t)ci * Cout + co) * 9 + tap];
      f1 = w[((size_t)(ci + 1) * Cout + co) * 9 + tap];
    } else {
      f0 = w[((size_t)co * Cin + ci) * 9 + tap];
      f1 = w[((size_t)co * Cin + ci + 1) * 9 + tap];
    }
  }
  unsigned short b0 = __builtin_bit_cast(unsigned short, (__bf16)f0);
  unsigned short b1 = __builtin_bit_cast(unsigned short, (__bf16)f1);
  wp[idx] = ((unsigned int)b1 << 16) | b0;
}

// ---------------------------------------------------------------------------
// 3x3 SAME conv, implicit GEMM. Input: NHWC bf16. Weights: packed A image.
// One wave -> 16 cout x 32 pixels. Taps OUTER, ci-chunks INNER: only one
// tap's mask/pointers live at a time; inner loop = pointer bumps only.
// ---------------------------------------------------------------------------
__global__ void __launch_bounds__(256)
conv3x3_wmma_kernel(const __bf16* __restrict__ xb,
                    const unsigned int* __restrict__ wp,
                    const float* __restrict__ bias,
                    float* __restrict__ out,
                    int Cin, int Cout, int H, int W, int coutTiles) {
  const int lane = threadIdx.x & 31;
  const int waveId = blockIdx.x * (blockDim.x >> 5) + (threadIdx.x >> 5);
  const int wTiles = W >> 5;
  const int pxTiles = H * wTiles;
  if (waveId >= pxTiles * coutTiles) return;   // wave-uniform

  const int coTile = waveId / pxTiles;
  const int px     = waveId - coTile * pxTiles;
  const int h  = px / wTiles;
  const int x0 = (px - h * wTiles) << 5;
  const int half = lane >> 4;
  const int nl   = lane & 15;
  const int KC   = Cin >> 5;
  const int cBase = half << 4;

  v8f acc0 = {}, acc1 = {};
  const unsigned int* wpt = wp + (size_t)coTile * KC * 9 * 256 + (lane << 3);

#pragma unroll
  for (int dy = -1; dy <= 1; ++dy) {
    const int yy = h + dy;
    const bool rowok = (yy >= 0) && (yy < H);
    const int yc = min(max(yy, 0), H - 1);
#pragma unroll
    for (int dx = -1; dx <= 1; ++dx) {
      const int tap = (dy + 1) * 3 + (dx + 1);
      const int xx0 = x0 + nl + dx;
      const int xx1 = xx0 + 16;
      const unsigned int m0 = (rowok && xx0 >= 0 && xx0 < W) ? 0xFFFFFFFFu : 0u;
      const unsigned int m1 = (rowok && xx1 >= 0 && xx1 < W) ? 0xFFFFFFFFu : 0u;
      const __bf16* b0p = xb + (size_t)(yc * W + min(max(xx0, 0), W - 1)) * Cin + cBase;
      const __bf16* b1p = xb + (size_t)(yc * W + min(max(xx1, 0), W - 1)) * Cin + cBase;
      const unsigned int* ap = wpt + tap * 256;
      for (int cc = 0; cc < KC; ++cc) {
        v16bf a = *(const v16bf*)ap;
        v8u u0 = *(const v8u*)b0p;
        v8u u1 = *(const v8u*)b1p;
        u0 &= m0;
        u1 &= m1;
        v16bf b0 = __builtin_bit_cast(v16bf, u0);
        v16bf b1 = __builtin_bit_cast(v16bf, u1);
        acc0 = __builtin_amdgcn_wmma_f32_16x16x32_bf16(false, a, false, b0,
                                                       (short)0, acc0, false, false);
        acc1 = __builtin_amdgcn_wmma_f32_16x16x32_bf16(false, a, false, b1,
                                                       (short)0, acc1, false, false);
        ap  += 9 * 256;   // next ci-chunk, same tap
        b0p += 32;
        b1p += 32;
      }
    }
  }
#pragma unroll
  for (int r = 0; r < 8; ++r) {
    int co = coTile * 16 + r + 8 * half;
    if (co < Cout) {
      float bv = bias[co];
      float* o = out + ((size_t)co * H + h) * W + x0 + nl;
      o[0]  = acc0[r] + bv;
      o[16] = acc1[r] + bv;
    }
  }
}

// ---------------------------------------------------------------------------
// PAC affinity kk
// ---------------------------------------------------------------------------
__global__ void __launch_bounds__(256)
pac_kk_kernel(const float* __restrict__ g, float* __restrict__ kk,
              int Cg, int Ho, int Wo) {
  const int pix = blockIdx.x;
  const int h = pix / Wo, wc = pix - h * Wo;
  float acc[9];
#pragma unroll
  for (int t = 0; t < 9; ++t) acc[t] = 0.f;

  for (int c = threadIdx.x; c < Cg; c += 256) {
    const float* gc = g + (size_t)c * Ho * Wo;
    float center = gc[(size_t)h * Wo + wc];
#pragma unroll
    for (int i = 0; i < 3; ++i) {
      int y = h + i - 1;
#pragma unroll
      for (int j = 0; j < 3; ++j) {
        int x = wc + j - 1;
        float nb = (y >= 0 && y < Ho && x >= 0 && x < Wo)
                       ? gc[(size_t)y * Wo + x] : 0.f;
        float d = nb - center;
        acc[i * 3 + j] += d * d;
      }
    }
  }
#pragma unroll
  for (int t = 0; t < 9; ++t)
#pragma unroll
    for (int off = 16; off > 0; off >>= 1)
      acc[t] += __shfl_xor(acc[t], off, 32);

  __shared__ float sh[8][9];
  const int wave = threadIdx.x >> 5, lane = threadIdx.x & 31;
  if (lane == 0)
#pragma unroll
    for (int t = 0; t < 9; ++t) sh[wave][t] = acc[t];
  __syncthreads();
  if (threadIdx.x < 9) {
    float s = 0.f;
#pragma unroll
    for (int w = 0; w < 8; ++w) s += sh[w][threadIdx.x];
    kk[(size_t)threadIdx.x * Ho * Wo + pix] = expf(-0.5f * s);
  }
}

// ---------------------------------------------------------------------------
// PAC transposed conv (stride2,pad1,outpad1). Input: low-res NHWC bf16.
// kk scaling + column-parity masking fused into v_pk_mul_bf16.
// Row parity is wave-uniform -> tap rows skipped without EXEC divergence.
// ---------------------------------------------------------------------------
__global__ void __launch_bounds__(256)
pac_wmma_kernel(const __bf16* __restrict__ xb,
                const float* __restrict__ kk,
                const unsigned int* __restrict__ wp,
                const float* __restrict__ bias,
                float* __restrict__ out,
                int Cin, int Cout, int H, int W,
                int Ho, int Wo, int coutTiles) {
  const int lane = threadIdx.x & 31;
  const int waveId = blockIdx.x * (blockDim.x >> 5) + (threadIdx.x >> 5);
  const int wTiles = Wo >> 5;
  const int pxTiles = Ho * wTiles;
  if (waveId >= pxTiles * coutTiles) return;   // wave-uniform

  const int coTile = waveId / pxTiles;
  const int px     = waveId - coTile * pxTiles;
  const int h  = px / wTiles;
  const int x0 = (px - h * wTiles) << 5;
  const int half = lane >> 4;
  const int nl   = lane & 15;
  const int KC   = Cin >> 5;
  const int cBase = half << 4;
  const int xcol0 = x0 + nl;

  v8f acc0 = {}, acc1 = {};
  const unsigned int* wpt = wp + (size_t)coTile * KC * 9 * 256 + (lane << 3);

#pragma unroll
  for (int i = 0; i < 3; ++i) {
    const int u = h + i - 1;                            // upsampled-grid row
    if (u < 0 || u > 2 * H - 2 || (u & 1)) continue;    // wave-uniform skip
    const int arow = u >> 1;
#pragma unroll
    for (int j = 0; j < 3; ++j) {
      const int tap = i * 3 + j;
      const float* kkrow = kk + ((size_t)tap * Ho + h) * Wo + xcol0;
      const int v0 = xcol0 + j - 1;                     // upsampled-grid col
      const int v1 = v0 + 16;
      const bool ok0 = (v0 >= 0) && (v0 <= 2 * W - 2) && ((v0 & 1) == 0);
      const bool ok1 = (v1 >= 0) && (v1 <= 2 * W - 2) && ((v1 & 1) == 0);
      const unsigned int km0 = ok0 ? bf16pair(kkrow[0])  : 0u;
      const unsigned int km1 = ok1 ? bf16pair(kkrow[16]) : 0u;
      const int bc0 = min(max(v0 >> 1, 0), W - 1);
      const int bc1 = min(max(v1 >> 1, 0), W - 1);
      const __bf16* b0p = xb + (size_t)(arow * W + bc0) * Cin + cBase;
      const __bf16* b1p = xb + (size_t)(arow * W + bc1) * Cin + cBase;
      const unsigned int* ap = wpt + tap * 256;
      for (int cc = 0; cc < KC; ++cc) {
        v16bf a = *(const v16bf*)ap;
        v8u u0 = *(const v8u*)b0p;
        v8u u1 = *(const v8u*)b1p;
#pragma unroll
        for (int q = 0; q < 8; ++q) {
          u0[q] = pk_mul_bf16(u0[q], km0);
          u1[q] = pk_mul_bf16(u1[q], km1);
        }
        v16bf b0 = __builtin_bit_cast(v16bf, u0);
        v16bf b1 = __builtin_bit_cast(v16bf, u1);
        acc0 = __builtin_amdgcn_wmma_f32_16x16x32_bf16(false, a, false, b0,
                                                       (short)0, acc0, false, false);
        acc1 = __builtin_amdgcn_wmma_f32_16x16x32_bf16(false, a, false, b1,
                                                       (short)0, acc1, false, false);
        ap  += 9 * 256;
        b0p += 32;
        b1p += 32;
      }
    }
  }
#pragma unroll
  for (int r = 0; r < 8; ++r) {
    int co = coTile * 16 + r + 8 * half;
    if (co < Cout) {
      float bv = bias[co];
      float* o = out + ((size_t)co * Ho + h) * Wo + x0 + nl;
      o[0]  = acc0[r] + bv;
      o[16] = acc1[r] + bv;
    }
  }
}

// ---------------------------------------------------------------------------
// Host-side orchestration
// ---------------------------------------------------------------------------
extern "C" void kernel_launch(void* const* d_in, const int* in_sizes, int n_in,
                              void* d_out, int out_size, void* d_ws, size_t ws_size,
                              hipStream_t stream) {
  const float* x      = (const float*)d_in[0];   // (256,64,64)
  const float* ef2    = (const float*)d_in[1];   // (128,128,128)
  const float* ef1    = (const float*)d_in[2];   // (64,256,256)
  const float* w_adj2 = (const float*)d_in[3];   // (256,128,3,3) OIHW
  const float* b_adj2 = (const float*)d_in[4];
  const float* w_adj1 = (const float*)d_in[5];   // (128,64,3,3)
  const float* b_adj1 = (const float*)d_in[6];
  const float* w_p16  = (const float*)d_in[7];   // (256,128,3,3) (Cin,Cout,3,3)
  const float* b_p16  = (const float*)d_in[8];
  const float* w_p20  = (const float*)d_in[9];   // (128,64,3,3)
  const float* b_p20  = (const float*)d_in[10];
  const float* wO     = (const float*)d_in[11];  // (3,64,3,3)
  const float* bO     = (const float*)d_in[12];
  float* outp = (float*)d_out;                   // (3,256,256)

  char* wsp = (char*)d_ws;
  auto alloc = [&](size_t bytes) {
    char* p = wsp;
    wsp += (bytes + 255) & ~(size_t)255;
    return p;
  };
  // f32 NCHW intermediates
  float* tA = (float*)alloc((size_t)256 * 64 * 64 * 4);
  float* g2 = (float*)alloc((size_t)256 * 128 * 128 * 4);
  float* k1 = (float*)alloc((size_t)9 * 128 * 128 * 4);
  float* p1 = (float*)alloc((size_t)128 * 128 * 128 * 4);
  float* g1 = (float*)alloc((size_t)128 * 256 * 256 * 4);
  float* k2 = (float*)alloc((size_t)9 * 256 * 256 * 4);
  float* p2 = (float*)alloc((size_t)64 * 256 * 256 * 4);
  // bf16 NHWC stagings
  __bf16* tAb  = (__bf16*)alloc((size_t)256 * 64 * 64 * 2);
  __bf16* ef2b = (__bf16*)alloc((size_t)128 * 128 * 128 * 2);
  __bf16* ef1b = (__bf16*)alloc((size_t)64 * 256 * 256 * 2);
  __bf16* p1b  = (__bf16*)alloc((size_t)128 * 128 * 128 * 2);
  __bf16* p2b  = (__bf16*)alloc((size_t)64 * 256 * 256 * 2);
  // packed bf16 A-fragment weights
  unsigned int* wpA2 = (unsigned int*)alloc((size_t)16 * 4 * 9 * 256 * 4);
  unsigned int* wpA1 = (unsigned int*)alloc((size_t)8 * 2 * 9 * 256 * 4);
  unsigned int* wpP1 = (unsigned int*)alloc((size_t)8 * 8 * 9 * 256 * 4);
  unsigned int* wpP2 = (unsigned int*)alloc((size_t)4 * 4 * 9 * 256 * 4);
  unsigned int* wpO  = (unsigned int*)alloc((size_t)1 * 2 * 9 * 256 * 4);

  // --- prep (tiny) ---------------------------------------------------------
  prepack_wmma_A<<<(16 * 4 * 9 * 256) / 256, 256, 0, stream>>>(w_adj2, wpA2, 128, 256, 16, 0);
  prepack_wmma_A<<<(8 * 2 * 9 * 256) / 256, 256, 0, stream>>>(w_adj1, wpA1, 64, 128, 8, 0);
  prepack_wmma_A<<<(8 * 8 * 9 * 256) / 256, 256, 0, stream>>>(w_p16, wpP1, 256, 128, 8, 1);
  prepack_wmma_A<<<(4 * 4 * 9 * 256) / 256, 256, 0, stream>>>(w_p20, wpP2, 128, 64, 4, 1);
  prepack_wmma_A<<<(1 * 2 * 9 * 256) / 256, 256, 0, stream>>>(wO, wpO, 64, 3, 1, 0);
  nchw_to_nhwc_bf16<<<(128 * 128 * 128) / 256, 256, 0, stream>>>(ef2, ef2b, 128, 128 * 128);
  nchw_to_nhwc_bf16<<<(64 * 256 * 256) / 256, 256, 0, stream>>>(ef1, ef1b, 64, 256 * 256);

  // --- stage 0: 2x inorm+residual on x; second pass emits bf16 NHWC --------
  inorm_res_kernel<<<256, 256, 0, stream>>>(x, tA, nullptr, 256, 64 * 64);
  inorm_res_kernel<<<256, 256, 0, stream>>>(tA, tA, tAb, 256, 64 * 64);

  // guide g2 = conv3x3(ef_lv2)  Cin=128 Cout=256 @128x128
  conv3x3_wmma_kernel<<<(128 * 4 * 16) / 8, 256, 0, stream>>>(
      ef2b, wpA2, b_adj2, g2, 128, 256, 128, 128, 16);
  pac_kk_kernel<<<128 * 128, 256, 0, stream>>>(g2, k1, 256, 128, 128);

  // PAC transpose #1: (256,64,64) -> (128,128,128)
  pac_wmma_kernel<<<(128 * 4 * 8) / 8, 256, 0, stream>>>(
      tAb, k1, wpP1, b_p16, p1, 256, 128, 64, 64, 128, 128, 8);
  inorm_res_kernel<<<128, 256, 0, stream>>>(p1, p1, nullptr, 128, 128 * 128);
  inorm_res_kernel<<<128, 256, 0, stream>>>(p1, p1, p1b, 128, 128 * 128);

  // guide g1 = conv3x3(ef_lv1)  Cin=64 Cout=128 @256x256
  conv3x3_wmma_kernel<<<(256 * 8 * 8) / 8, 256, 0, stream>>>(
      ef1b, wpA1, b_adj1, g1, 64, 128, 256, 256, 8);
  pac_kk_kernel<<<256 * 256, 256, 0, stream>>>(g1, k2, 128, 256, 256);

  // PAC transpose #2: (128,128,128) -> (64,256,256)
  pac_wmma_kernel<<<(256 * 8 * 4) / 8, 256, 0, stream>>>(
      p1b, k2, wpP2, b_p20, p2, 128, 64, 128, 128, 256, 256, 4);
  inorm_res_kernel<<<64, 256, 0, stream>>>(p2, p2, nullptr, 64, 256 * 256);
  inorm_res_kernel<<<64, 256, 0, stream>>>(p2, p2, p2b, 64, 256 * 256);

  // final conv: Cin=64 Cout=3 @256x256 (A image zero-padded to 16 cout)
  conv3x3_wmma_kernel<<<(256 * 8 * 1) / 8, 256, 0, stream>>>(
      p2b, wpO, bO, outp, 64, 3, 256, 256, 1);
}
